// DRA_40072044872030
// MI455X (gfx1250) — matrix operand order repaired and checked by reading, not compiled, not run on hardware
//
#include <hip/hip_runtime.h>
#include <hip/hip_bf16.h>

typedef __attribute__((ext_vector_type(16))) __bf16 v16bf;
typedef __attribute__((ext_vector_type(8)))  __bf16 v8bf;
typedef __attribute__((ext_vector_type(8)))  float  v8f;

#define BATCH   16
#define CIN     200
#define HWDIM   56
#define HW      3136          // 56*56
#define NPIX    50176         // 16*3136
#define KTOT    1800          // 200*9
#define KSTEPS  57            // ceil(1800/32)
#define KPAD    1824          // 57*32
#define KP      1832          // LDS row stride (halves), padded for bank spread
#define MTILES  16            // 256 padded channels -> uniform 2 tiles per wave
#define TOPK    313           // int(3136*0.1)

#define A_FRAG_BYTES   1024                    // 32 lanes * 32 B
#define A_TILE_BYTES   (KSTEPS * A_FRAG_BYTES) // bytes per M-tile (57 frags)
#define A_SECOND_OFF   (8 * A_TILE_BYTES)      // wave's 2nd M-tile: mt+8

// ---------------------------------------------------------------- prep: diff
__global__ __launch_bounds__(256) void DRA_diff_kernel(
    const float* __restrict__ x, const float* __restrict__ ref,
    __bf16* __restrict__ out, int n) {
  for (int i = blockIdx.x * blockDim.x + threadIdx.x; i < n;
       i += gridDim.x * blockDim.x) {
    out[i] = (__bf16)(ref[i] - x[i]);
  }
}

// ------------------------------------------------- prep: fold BN into scale/shift
__global__ __launch_bounds__(256) void DRA_params_kernel(
    const float* __restrict__ conv_b, const float* __restrict__ gamma,
    const float* __restrict__ beta,   const float* __restrict__ mean,
    const float* __restrict__ var,    const float* __restrict__ score_w,
    float* __restrict__ scale, float* __restrict__ shift, float* __restrict__ scw) {
  int t = threadIdx.x;  // one block of 256
  float sc = 0.f, sh = 0.f, sw = 0.f;
  if (t < CIN) {
    float inv = rsqrtf(var[t] + 1e-5f);
    sc = gamma[t] * inv;
    sh = sc * (conv_b[t] - mean[t]) + beta[t];
    sw = score_w[t];
  }
  scale[t] = sc; shift[t] = sh; scw[t] = sw;
}

// ----------------------------- prep: pack conv weights into WMMA A-fragment order
// A is M(16) x K(32) bf16 per fragment. Lane L holds row M = L%16;
// halves 0..7 -> K = 8*(L/16)+j ; halves 8..15 -> K = 16 + 8*(L/16) + (j-8).
__global__ __launch_bounds__(256) void DRA_packA_kernel(
    const float* __restrict__ w, __bf16* __restrict__ Apk) {
  const int total = MTILES * KSTEPS * 32 * 16;
  for (int i = blockIdx.x * blockDim.x + threadIdx.x; i < total;
       i += gridDim.x * blockDim.x) {
    int j    = i & 15;
    int lane = (i >> 4) & 31;
    int frag = i >> 9;               // mt*KSTEPS + ks
    int ks   = frag % KSTEPS;
    int mt   = frag / KSTEPS;
    int g    = lane >> 4;
    int m    = lane & 15;
    int k    = ks * 32 + (j < 8 ? 8 * g + j : 16 + 8 * g + (j - 8));
    int ch   = mt * 16 + m;
    float v  = 0.f;
    if (ch < CIN && k < KTOT) {
      int c  = k / 9;
      int kk = k - c * 9;            // ky*3+kx
      v = w[(ch * CIN + c) * 9 + kk];
    }
    Apk[i] = (__bf16)v;
  }
}

// ---------------------- main: implicit-GEMM 3x3 conv + BN + ReLU + 1x1 score conv
// One block per 16-pixel tile. 8 waves; wave w owns M-tiles {w, w+8} (uniform).
__global__ __launch_bounds__(256) void DRA_convgemm_kernel(
    const __bf16* __restrict__ dbf,  const __bf16* __restrict__ Apk,
    const float* __restrict__ scale, const float* __restrict__ shift,
    const float* __restrict__ scw,   const float* __restrict__ score_b,
    float* __restrict__ scores) {
  __shared__ __bf16 sB[16 * KP];
  __shared__ float  partial[16];

  const int tid  = threadIdx.x;
  const int tile = blockIdx.x;       // 0..3135 ; each tile stays inside one image
  const int gp0  = tile * 16;

  if (tid < 16) partial[tid] = 0.f;

  // ---- stage im2col B tile into LDS: [n=0..15][k=0..KP), k-contiguous
  for (int idx = tid; idx < 16 * KP; idx += 256) {
    int n = idx / KP;
    int k = idx - n * KP;
    __bf16 v = (__bf16)0.0f;
    if (k < KTOT) {
      int gp  = gp0 + n;
      int b   = gp / HW;
      int rem = gp - b * HW;
      int y   = rem / HWDIM;
      int x   = rem - y * HWDIM;
      int c   = k / 9;
      int kk  = k - c * 9;
      int ky  = kk / 3;
      int kx  = kk - ky * 3;
      int yy  = y + ky - 1;
      int xx  = x + kx - 1;
      if (yy >= 0 && yy < HWDIM && xx >= 0 && xx < HWDIM)
        v = dbf[((b * CIN + c) * HWDIM + yy) * HWDIM + xx];
    }
    sB[n * KP + k] = v;
  }
  __syncthreads();

  const int wave = tid >> 5;
  const int lane = tid & 31;
  const int r    = lane & 15;
  const int g    = lane >> 4;

  const int mt0 = wave;              // 0..7
  const int mt1 = wave + 8;          // 8..15 (padded channels: always valid)

  v8f acc0 = {0.f, 0.f, 0.f, 0.f, 0.f, 0.f, 0.f, 0.f};
  v8f acc1 = {0.f, 0.f, 0.f, 0.f, 0.f, 0.f, 0.f, 0.f};

  union BFrag { v16bf v; v8bf h[2]; };

  // incremental byte offsets: A advances 1024 B / k-step; second tile is a
  // constant displacement that folds into the load's immediate offset.
  const char* Ab   = (const char*)Apk;
  int         aoff = mt0 * A_TILE_BYTES + lane * 32;
  const __bf16* bp = &sB[r * KP + 8 * g];

#pragma unroll 3
  for (int ks = 0; ks < KSTEPS; ++ks) {
    BFrag fb;
    fb.h[0] = *(const v8bf*)(bp);
    fb.h[1] = *(const v8bf*)(bp + 16);

    const v16bf a0 = *(const v16bf*)(Ab + aoff);
    const v16bf a1 = *(const v16bf*)(Ab + aoff + A_SECOND_OFF);
    acc0 = __builtin_amdgcn_wmma_f32_16x16x32_bf16(
        false, a0, false, fb.v, (short)0, acc0, false, false);
    acc1 = __builtin_amdgcn_wmma_f32_16x16x32_bf16(
        false, a1, false, fb.v, (short)0, acc1, false, false);

    aoff += A_FRAG_BYTES;
    bp   += 32;
  }

  // ---- epilogue: fold BN (scale/shift), ReLU, dot with score_w; reduce per pixel
  // C layout: vgpr rr, lane L: channel row = rr + 8*(L/16), pixel N = L%16
  float lsum = 0.f;
#pragma unroll
  for (int rr = 0; rr < 8; ++rr) {
    int ch  = mt0 * 16 + rr + 8 * g;
    float h = fmaxf(scale[ch] * acc0[rr] + shift[ch], 0.f);
    lsum += scw[ch] * h;
  }
#pragma unroll
  for (int rr = 0; rr < 8; ++rr) {
    int ch  = mt1 * 16 + rr + 8 * g;
    float h = fmaxf(scale[ch] * acc1[rr] + shift[ch], 0.f);
    lsum += scw[ch] * h;
  }
  atomicAdd(&partial[r], lsum);   // ds_add_f32
  __syncthreads();

  if (tid < 16) {
    float s = partial[tid] + score_b[0];
    scores[gp0 + tid] = fabsf(s);   // gp0+tid == b*HW + pos
  }
}

// ---------------------------------------- top-k(313) mean per image via bitonic
__global__ __launch_bounds__(256) void DRA_topk_kernel(
    const float* __restrict__ scores, float* __restrict__ out) {
  __shared__ float s[4096];
  __shared__ float red[256];
  const int row = blockIdx.x;
  const int tid = threadIdx.x;

  for (int i = tid; i < 4096; i += 256)
    s[i] = (i < HW) ? scores[row * HW + i] : -1.0f;   // scores are |.| >= 0
  __syncthreads();

  for (int ksz = 2; ksz <= 4096; ksz <<= 1) {
    for (int j = ksz >> 1; j > 0; j >>= 1) {
      for (int i = tid; i < 4096; i += 256) {
        int ixj = i ^ j;
        if (ixj > i) {
          float a = s[i], b = s[ixj];
          bool up = ((i & ksz) == 0);     // ascending region
          if (up ? (a > b) : (b > a)) { s[i] = b; s[ixj] = a; }
        }
      }
      __syncthreads();
    }
  }

  float sum = 0.f;
  for (int i = 4096 - TOPK + tid; i < 4096; i += 256) sum += s[i];
  red[tid] = sum;
  __syncthreads();
  for (int st = 128; st > 0; st >>= 1) {
    if (tid < st) red[tid] += red[tid + st];
    __syncthreads();
  }
  if (tid == 0) out[row] = red[0] / (float)TOPK;
}

// -----------------------------------------------------------------------------
extern "C" void kernel_launch(void* const* d_in, const int* in_sizes, int n_in,
                              void* d_out, int out_size, void* d_ws, size_t ws_size,
                              hipStream_t stream) {
  const float* feature     = (const float*)d_in[0];
  const float* ref_feature = (const float*)d_in[1];
  // d_in[2..9]: attention-module params — provably unused (softmax over size-1
  // axis is identically 1.0, so the attention module is an exact identity).
  const float* comp_conv_w = (const float*)d_in[10];
  const float* comp_conv_b = (const float*)d_in[11];
  const float* bn_gamma    = (const float*)d_in[12];
  const float* bn_beta     = (const float*)d_in[13];
  const float* bn_mean     = (const float*)d_in[14];
  const float* bn_var      = (const float*)d_in[15];
  const float* score_w     = (const float*)d_in[16];
  const float* score_b     = (const float*)d_in[17];
  float* out = (float*)d_out;

  char* ws = (char*)d_ws;
  size_t o = 0;
  __bf16* dbf = (__bf16*)(ws + o); o += (size_t)BATCH * CIN * HW * 2;        // 20,070,400
  __bf16* Apk = (__bf16*)(ws + o); o += (size_t)MTILES * KSTEPS * 512 * 2;   //    933,888
  float* scale = (float*)(ws + o); o += 256 * 4;
  float* shift = (float*)(ws + o); o += 256 * 4;
  float* scw   = (float*)(ws + o); o += 256 * 4;
  float* scores = (float*)(ws + o); o += (size_t)NPIX * 4;                   //    200,704

  const int ndiff = BATCH * CIN * HW;   // 10,035,200
  DRA_diff_kernel<<<4096, 256, 0, stream>>>(feature, ref_feature, dbf, ndiff);
  DRA_params_kernel<<<1, 256, 0, stream>>>(comp_conv_b, bn_gamma, bn_beta,
                                           bn_mean, bn_var, score_w,
                                           scale, shift, scw);
  DRA_packA_kernel<<<(MTILES * KSTEPS * 512 + 255) / 256, 256, 0, stream>>>(
      comp_conv_w, Apk);
  DRA_convgemm_kernel<<<NPIX / 16, 256, 0, stream>>>(dbf, Apk, scale, shift,
                                                     scw, score_b, scores);
  DRA_topk_kernel<<<BATCH, 256, 0, stream>>>(scores, out);
}
